// SwitchGNN_20942260535353
// MI455X (gfx1250) — matrix operand order repaired and checked by reading, not compiled, not run on hardware
//
#include <hip/hip_runtime.h>
#include <hip/hip_bf16.h>

typedef __attribute__((ext_vector_type(16))) _Float16 v16h;
typedef __attribute__((ext_vector_type(8)))  float    v8f;

#define DDIM 128
#define LDSW_STRIDE 130   // pad 128 -> 130 halves to kill LDS bank conflicts

// ---------------------------------------------------------------------------
// Zero the per-type accumulator (a[N*D] followed by cnt[N], contiguous).
// ---------------------------------------------------------------------------
__global__ void zero_kernel(float* __restrict__ p, int n) {
    int i = blockIdx.x * blockDim.x + threadIdx.x;
    if (i < n) p[i] = 0.0f;
}

// ---------------------------------------------------------------------------
// Scatter stage: one wave (32 lanes) per edge. Each lane owns 4 of the 128
// features (float4 load from x, 4x global_atomic_add_f32 into a[dst]).
// Lane 0 bumps the edge count. x (51 MB) is L2-resident on MI455X (192 MB L2),
// so these gathers hit L2, not HBM.
// ---------------------------------------------------------------------------
__global__ void __launch_bounds__(256)
scatter_kernel(const float* __restrict__ x,
               const long long* __restrict__ src,
               const long long* __restrict__ dst,
               float* __restrict__ a, float* __restrict__ cnt, int E) {
    int wave = (blockIdx.x * blockDim.x + threadIdx.x) >> 5;
    int lane = threadIdx.x & 31;
    if (wave >= E) return;
    int s = (int)src[wave];
    int d = (int)dst[wave];
    const float4 v = *(const float4*)(x + (size_t)s * DDIM + lane * 4);
    float* ad = a + (size_t)d * DDIM + lane * 4;
    atomicAdd(ad + 0, v.x);
    atomicAdd(ad + 1, v.y);
    atomicAdd(ad + 2, v.z);
    atomicAdd(ad + 3, v.w);
    if (lane == 0) atomicAdd(&cnt[d], 1.0f);
}

// ---------------------------------------------------------------------------
// GEMM stage: out += (1/7) * [ (a / max(cnt,1)) @ W_t + b_t*(cnt>0) ]
//   A = a/cnt : [N x 128] fp32 -> f16 on the fly
//   B = W_t   : [128 x 128] fp32 -> f16, staged transposed in LDS
// Block = 256 threads = 8 waves; each wave computes a 16x128 output tile
// with 8 f32 accumulators, 4 K-steps of v_wmma_f32_16x16x32_f16 per N-tile.
// ---------------------------------------------------------------------------
__global__ void __launch_bounds__(256)
gemm_acc_kernel(const float* __restrict__ a, const float* __restrict__ cnt,
                const float* __restrict__ Wt, const float* __restrict__ bt,
                float* __restrict__ out, int N, int init) {
    __shared__ _Float16 Wlds[DDIM * LDSW_STRIDE];

    int tid = threadIdx.x;
    // Load W_t transposed: Wlds[f][d] = (f16) Wt[d*128 + f]  (coalesced reads)
    for (int idx = tid; idx < DDIM * DDIM; idx += 256) {
        int d = idx >> 7;
        int f = idx & 127;
        Wlds[f * LDSW_STRIDE + d] = (_Float16)Wt[idx];
    }
    __syncthreads();

    int wave  = tid >> 5;
    int lane  = tid & 31;
    int mtile = blockIdx.x * 8 + wave;
    int m0    = mtile * 16;
    if (m0 >= N) return;                 // after the only barrier: safe

    int lhalf = lane >> 4;               // which K-half this lane covers
    int l16   = lane & 15;

    // A row this lane feeds (ISA: A lanes 0-15 and 16-31 both carry M=0..15)
    int  mA    = m0 + l16;
    bool mAok  = (mA < N);
    const float* arow = a + (size_t)(mAok ? mA : 0) * DDIM;
    float cA   = mAok ? cnt[mA] : 0.0f;
    float inv  = mAok ? (1.0f / fmaxf(cA, 1.0f)) : 0.0f;

    v8f acc[8] = {};                     // 8 N-tiles of 16x16 f32

    for (int ks = 0; ks < 4; ++ks) {
        int kbase = ks * 32;
        // A fragment, 16 halves / lane. Element e -> K index:
        //   k = kbase + lhalf*16 + e
        v16h afrag;
#pragma unroll
        for (int p = 0; p < 8; ++p) {
            int k = kbase + lhalf * 16 + p * 2;
            float2 v = *(const float2*)(arow + k);   // b64 load, L2-hot
            afrag[2 * p]     = (_Float16)(v.x * inv);
            afrag[2 * p + 1] = (_Float16)(v.y * inv);
        }
#pragma unroll
        for (int ft = 0; ft < 8; ++ft) {
            int f = ft * 16 + l16;       // B: lane carries column n
            const _Float16* wl = &Wlds[f * LDSW_STRIDE + kbase + lhalf * 16];
            v16h bfrag;
#pragma unroll
            for (int e = 0; e < 16; ++e) bfrag[e] = wl[e];  // 32B contiguous ds reads
            acc[ft] = __builtin_amdgcn_wmma_f32_16x16x32_f16(
                false, afrag, false, bfrag, (short)0, acc[ft], false, false);
        }
    }

    // C/D layout: lane l16 = column; VGPR r -> row r + lhalf*8
    const float scale = 1.0f / 7.0f;
    float cm[8];
#pragma unroll
    for (int r = 0; r < 8; ++r) {
        int m = m0 + r + lhalf * 8;
        cm[r] = (m < N && cnt[m] > 0.0f) ? 1.0f : 0.0f;
    }
#pragma unroll
    for (int ft = 0; ft < 8; ++ft) {
        int f = ft * 16 + l16;
        float bterm = bt[f] * scale;
#pragma unroll
        for (int r = 0; r < 8; ++r) {
            int m = m0 + r + lhalf * 8;
            if (m < N) {
                float val = acc[ft][r] * scale + bterm * cm[r];
                size_t oi = (size_t)m * DDIM + f;
                if (init) out[oi] = val;
                else      out[oi] += val;
            }
        }
    }
}

// ---------------------------------------------------------------------------
// Host launcher: per edge type t: zero accumulator -> scatter x -> WMMA GEMM
// accumulate into out. All on `stream`; graph-capture safe.
// ---------------------------------------------------------------------------
extern "C" void kernel_launch(void* const* d_in, const int* in_sizes, int n_in,
                              void* d_out, int out_size, void* d_ws, size_t ws_size,
                              hipStream_t stream) {
    const float*     x  = (const float*)d_in[0];
    const long long* ei = (const long long*)d_in[1];   // [T,2,E] int64
    const float*     W  = (const float*)d_in[2];       // [T,128,128]
    const float*     b  = (const float*)d_in[3];       // [T,128]
    float*           out = (float*)d_out;

    const int T = 7;
    const int N = in_sizes[0] / DDIM;                  // 100000
    const int E = in_sizes[1] / (T * 2);               // 250000

    float* a   = (float*)d_ws;                         // [N,128] accumulator
    float* cnt = a + (size_t)N * DDIM;                 // [N] edge counts (contiguous)

    const int zn = N * DDIM + N;
    dim3 zgrid((zn + 255) / 256);
    dim3 sgrid(((size_t)E * 32 + 255) / 256);
    dim3 ggrid(((N + 15) / 16 + 7) / 8);

    for (int t = 0; t < T; ++t) {
        zero_kernel<<<zgrid, 256, 0, stream>>>(a, zn);
        const long long* src = ei + (size_t)t * 2 * E;
        const long long* dst = src + E;
        scatter_kernel<<<sgrid, 256, 0, stream>>>(x, src, dst, a, cnt, E);
        gemm_acc_kernel<<<ggrid, 256, 0, stream>>>(
            a, cnt, W + (size_t)t * DDIM * DDIM, b + (size_t)t * DDIM,
            out, N, (t == 0) ? 1 : 0);
    }
}
